// TGATBaseline_40072044872163
// MI455X (gfx1250) — matrix operand order repaired and checked by reading, not compiled, not run on hardware
//
#include <hip/hip_runtime.h>
#include <math.h>

typedef __attribute__((ext_vector_type(16))) __bf16          v16bf;
typedef __attribute__((ext_vector_type(8)))  float           v8f;

#define HO_C 128
#define NNODES 50000

// ---------- helpers ----------
__device__ __forceinline__ unsigned short f2bf(float f) {
    // hardware f32 -> bf16 (RNE) conversion; backend emits v_cvt
    __bf16 h = (__bf16)f;
    return __builtin_bit_cast(unsigned short, h);
}

__device__ __forceinline__ void atomicMaxF(float* addr, float val) {
    if (!(val < 0.0f)) atomicMax((int*)addr, __float_as_int(val));
    else               atomicMin((unsigned int*)addr, __float_as_uint(val));
}

__device__ __forceinline__ void atomicAddF(float* addr, float val) {
    unsafeAtomicAdd(addr, val);   // hw global_atomic_add_f32
}

// ---------- K0: Wg[i][h*32+j] = sum_d wq[i][h*32+d]*wt[j][h*32+d]; bg from bq ----------
__global__ void make_wg_kernel(const float* __restrict__ wq, const float* __restrict__ bq,
                               const float* __restrict__ wt, float* __restrict__ Wg,
                               float* __restrict__ bg, int K) {
    int idx = blockIdx.x * blockDim.x + threadIdx.x;
    int total = (K + 1) * HO_C;
    if (idx >= total) return;
    int i  = idx >> 7;
    int hj = idx & 127;
    int h  = hj >> 5;
    int j  = hj & 31;
    const float* arow = (i < K) ? (wq + (size_t)i * HO_C) : bq;
    const float* wrow = wt + (size_t)j * HO_C + h * 32;
    float acc = 0.0f;
    #pragma unroll
    for (int d = 0; d < 32; ++d) acc += arow[h * 32 + d] * wrow[d];
    if (i < K) Wg[(size_t)i * HO_C + hj] = acc;
    else       bg[hj] = acc;
}

// ---------- K1: Y = X @ W + b  (4 outputs selected by blockIdx.y), bf16 WMMA ----------
__global__ __launch_bounds__(256) void gemm_qkvg_kernel(
    const float* __restrict__ X, int n_rows, int K,
    const float* __restrict__ Wq, const float* __restrict__ Wk,
    const float* __restrict__ Wv, const float* __restrict__ Wg,
    const float* __restrict__ bq, const float* __restrict__ bk,
    const float* __restrict__ bv, const float* __restrict__ bg,
    const float* __restrict__ bt_extra,   // added to k bias (folds te bias)
    float* __restrict__ Yq, float* __restrict__ Yk,
    float* __restrict__ Yv, float* __restrict__ Yg)
{
    const int which = blockIdx.y;
    const float* W = (which == 0) ? Wq : (which == 1) ? Wk : (which == 2) ? Wv : Wg;
    const float* b = (which == 0) ? bq : (which == 1) ? bk : (which == 2) ? bv : bg;
    float*       Y = (which == 0) ? Yq : (which == 1) ? Yk : (which == 2) ? Yv : Yg;

    // Stage W (K x 128, f32) into LDS as bf16 in B-fragment layout:
    // fragment (kchunk, ntile, lane) = 16 contiguous bf16 (32B) -> one ds_load_b128 pair.
    __shared__ __attribute__((aligned(32))) unsigned short ldsB[128 * 128];
    const int tid = threadIdx.x;
    const int total = K * HO_C;
    for (int idx = tid; idx < total; idx += 256) {
        int k = idx >> 7, n = idx & 127;
        int kchunk = k >> 5, kk = k & 31;
        int lane   = ((kk >> 4) << 4) | (n & 15);
        int j      = kk & 15;
        int ntile  = n >> 4;
        ldsB[(((kchunk * 8 + ntile) * 32 + lane) << 4) + j] = f2bf(W[idx]);
    }
    __syncthreads();

    const int wave = tid >> 5;
    const int lane = tid & 31;
    const int m0   = blockIdx.x * 128 + wave * 16;   // 8 waves x 16 rows
    if (m0 >= n_rows) return;                        // wave-uniform: EXEC stays full

    const int mrow = m0 + (lane & 15);
    const int koff = (lane >> 4) * 16;               // A-fragment K offset per half-wave

    v8f acc[8];
    #pragma unroll
    for (int t = 0; t < 8; ++t) {
        v8f z = {0.f, 0.f, 0.f, 0.f, 0.f, 0.f, 0.f, 0.f};
        acc[t] = z;
    }

    for (int kc = 0; kc < K; kc += 32) {
        // A fragment (16x32 bf16): lane row = lane&15, K = koff+{0..15}
        const float* xp = X + (size_t)mrow * K + kc + koff;
        v16bf av;
        #pragma unroll
        for (int j = 0; j < 8; ++j) av[j]     = (__bf16)xp[j];
        #pragma unroll
        for (int j = 0; j < 8; ++j) av[8 + j] = (__bf16)xp[8 + j];

        const int kchunk = kc >> 5;
        #pragma unroll
        for (int nt = 0; nt < 8; ++nt) {
            v16bf bfrag = *(const v16bf*)&ldsB[(((kchunk * 8 + nt) * 32 + lane) << 4)];
            acc[nt] = __builtin_amdgcn_wmma_f32_16x16x32_bf16(
                false, av, false, bfrag, (short)0, acc[nt], false, false);
        }
    }

    // C layout: element i -> row m0 + i + (lane>=16 ? 8 : 0), col = ntile*16 + (lane&15)
    const int col0  = lane & 15;
    const int rbase = m0 + ((lane >> 4) << 3);
    #pragma unroll
    for (int nt = 0; nt < 8; ++nt) {
        int col = nt * 16 + col0;
        float bias = b[col];
        if (which == 1) bias += bt_extra[col];
        #pragma unroll
        for (int i = 0; i < 8; ++i)
            Y[(size_t)(rbase + i) * HO_C + col] = acc[nt][i] + bias;
    }
}

// ---------- init: m=-inf, s=0, xacc=0 ----------
__global__ void init_layer_kernel(float* __restrict__ m, float* __restrict__ s,
                                  float* __restrict__ xacc, int n4, int n128) {
    int i = blockIdx.x * blockDim.x + threadIdx.x;
    if (i < n4) { m[i] = __int_as_float(0xff800000); s[i] = 0.0f; }
    if (i < n128) xacc[i] = 0.0f;
}

// ---------- K2: per-edge alpha + segment max (one wave per edge) ----------
__global__ __launch_bounds__(256) void edge_alpha_kernel(
    const int* __restrict__ ei, const float* __restrict__ ea,
    const float* __restrict__ q, const float* __restrict__ kf,
    const float* __restrict__ g, float* __restrict__ alpha,
    float* __restrict__ m, int E_)
{
    int e = blockIdx.x * 8 + (threadIdx.x >> 5);
    if (e >= E_) return;
    int lane = threadIdx.x & 31;
    int s = ei[e];
    int d = ei[E_ + e];
    float t = ea[2 * e + 1];

    // tfeat[lane]: even lanes sin(t*div[l/2]), odd lanes cos; div = exp(2i * -ln(1e4)/32)
    float arg = (float)(lane & ~1) * (-0.28782313662242554f);
    float ang = t * __expf(arg);
    float tf  = (lane & 1) ? __cosf(ang) : __sinf(ang);

    const float* qd = q  + (size_t)d * HO_C;
    const float* ks = kf + (size_t)s * HO_C;
    const float* gd = g  + (size_t)d * HO_C;

    float p[4];
    #pragma unroll
    for (int h = 0; h < 4; ++h) {
        int c = h * 32 + lane;
        p[h] = qd[c] * ks[c] + tf * gd[c];
    }
    #pragma unroll
    for (int off = 16; off > 0; off >>= 1) {
        #pragma unroll
        for (int h = 0; h < 4; ++h) p[h] += __shfl_xor(p[h], off, 32);
    }
    if (lane < 4) {
        float a = p[lane] * 0.17677669529663687f;   // 1/sqrt(32)
        alpha[(size_t)e * 4 + lane] = a;
        atomicMaxF(&m[d * 4 + lane], a);
    }
}

// ---------- K3: ex = exp(alpha - m[dst]); segment sum ----------
__global__ void edge_exp_kernel(const int* __restrict__ ei, float* __restrict__ alpha,
                                const float* __restrict__ m, float* __restrict__ s, int E_) {
    int i = blockIdx.x * blockDim.x + threadIdx.x;
    if (i >= E_ * 4) return;
    int e = i >> 2, h = i & 3;
    int d = ei[E_ + e];
    float ex = __expf(alpha[i] - m[d * 4 + h]);
    alpha[i] = ex;
    atomicAddF(&s[d * 4 + h], ex);
}

// ---------- K4: weighted scatter (one wave per edge, coalesced atomics) ----------
__global__ __launch_bounds__(256) void edge_scatter_kernel(
    const int* __restrict__ ei, const float* __restrict__ alpha,
    const float* __restrict__ s, const float* __restrict__ v,
    float* __restrict__ xacc, int E_)
{
    int e = blockIdx.x * 8 + (threadIdx.x >> 5);
    if (e >= E_) return;
    int lane = threadIdx.x & 31;
    int sn = ei[e];
    int d  = ei[E_ + e];
    #pragma unroll
    for (int h = 0; h < 4; ++h) {
        float a = alpha[(size_t)e * 4 + h] / (s[d * 4 + h] + 1e-16f);
        int c = h * 32 + lane;
        atomicAddF(&xacc[(size_t)d * HO_C + c], v[(size_t)sn * HO_C + c] * a);
    }
}

// ---------- ReLU in place ----------
__global__ void relu_kernel(float* __restrict__ x, int n) {
    int i = blockIdx.x * blockDim.x + threadIdx.x;
    if (i < n) x[i] = fmaxf(x[i], 0.0f);
}

// ---------- K6: out[e] = (x[src]+x[dst]) . wc + bc  (one wave per edge) ----------
__global__ __launch_bounds__(256) void edge_out_kernel(
    const int* __restrict__ ei, const float* __restrict__ x,
    const float* __restrict__ wc, const float* __restrict__ bc,
    float* __restrict__ out, int E_)
{
    int e = blockIdx.x * 8 + (threadIdx.x >> 5);
    if (e >= E_) return;
    int lane = threadIdx.x & 31;
    int s = ei[e];
    int d = ei[E_ + e];
    float acc = 0.0f;
    #pragma unroll
    for (int k = 0; k < 4; ++k) {
        int c = k * 32 + lane;
        acc += (x[(size_t)s * HO_C + c] + x[(size_t)d * HO_C + c]) * wc[c];
    }
    #pragma unroll
    for (int off = 16; off > 0; off >>= 1) acc += __shfl_xor(acc, off, 32);
    if (lane == 0) out[e] = acc + bc[0];
}

// ============================ launch ============================
extern "C" void kernel_launch(void* const* d_in, const int* in_sizes, int n_in,
                              void* d_out, int out_size, void* d_ws, size_t ws_size,
                              hipStream_t stream) {
    const int N = NNODES;
    const int E = in_sizes[0] / 2;

    const int*   ei  = (const int*)  d_in[0];
    const float* ea  = (const float*)d_in[1];
    const float* emb = (const float*)d_in[3];
    const float* wq1 = (const float*)d_in[4];  const float* bq1 = (const float*)d_in[5];
    const float* wk1 = (const float*)d_in[6];  const float* bk1 = (const float*)d_in[7];
    const float* wv1 = (const float*)d_in[8];  const float* bv1 = (const float*)d_in[9];
    const float* wt1 = (const float*)d_in[10]; const float* bt1 = (const float*)d_in[11];
    const float* wq2 = (const float*)d_in[12]; const float* bq2 = (const float*)d_in[13];
    const float* wk2 = (const float*)d_in[14]; const float* bk2 = (const float*)d_in[15];
    const float* wv2 = (const float*)d_in[16]; const float* bv2 = (const float*)d_in[17];
    const float* wt2 = (const float*)d_in[18]; const float* bt2 = (const float*)d_in[19];
    const float* wc  = (const float*)d_in[20]; const float* bc  = (const float*)d_in[21];
    float* out = (float*)d_out;

    // workspace layout (floats)
    float* ws = (float*)d_ws;
    float* q   = ws;
    float* kb  = q  + (size_t)N * HO_C;
    float* vb  = kb + (size_t)N * HO_C;
    float* gb  = vb + (size_t)N * HO_C;
    float* xb  = gb + (size_t)N * HO_C;     // layer accumulator / activations
    float* al  = xb + (size_t)N * HO_C;     // E*4 alpha/ex
    float* mb  = al + (size_t)E * 4;
    float* sb  = mb + (size_t)N * 4;
    float* Wg  = sb + (size_t)N * 4;        // 128*128
    float* bg  = Wg + 128 * 128;            // 128

    const dim3 blk(256);
    const int gGemmX = (N + 127) / 128;
    const int gEdgeW = (E + 7) / 8;         // one wave per edge, 8 waves/block
    const int gNode  = (N * HO_C + 255) / 256;

    // ---------------- layer 1 (K = 64) ----------------
    make_wg_kernel<<<(((64 + 1) * HO_C) + 255) / 256, blk, 0, stream>>>(wq1, bq1, wt1, Wg, bg, 64);
    gemm_qkvg_kernel<<<dim3(gGemmX, 4), blk, 0, stream>>>(
        emb, N, 64, wq1, wk1, wv1, Wg, bq1, bk1, bv1, bg, bt1, q, kb, vb, gb);
    init_layer_kernel<<<gNode, blk, 0, stream>>>(mb, sb, xb, N * 4, N * HO_C);
    edge_alpha_kernel<<<gEdgeW, blk, 0, stream>>>(ei, ea, q, kb, gb, al, mb, E);
    edge_exp_kernel<<<(E * 4 + 255) / 256, blk, 0, stream>>>(ei, al, mb, sb, E);
    edge_scatter_kernel<<<gEdgeW, blk, 0, stream>>>(ei, al, sb, vb, xb, E);
    relu_kernel<<<gNode, blk, 0, stream>>>(xb, N * HO_C);

    // ---------------- layer 2 (K = 128) ----------------
    make_wg_kernel<<<(((128 + 1) * HO_C) + 255) / 256, blk, 0, stream>>>(wq2, bq2, wt2, Wg, bg, 128);
    gemm_qkvg_kernel<<<dim3(gGemmX, 4), blk, 0, stream>>>(
        xb, N, 128, wq2, wk2, wv2, Wg, bq2, bk2, bv2, bg, bt2, q, kb, vb, gb);
    init_layer_kernel<<<gNode, blk, 0, stream>>>(mb, sb, xb, N * 4, N * HO_C);  // xb reusable after gemm read
    edge_alpha_kernel<<<gEdgeW, blk, 0, stream>>>(ei, ea, q, kb, gb, al, mb, E);
    edge_exp_kernel<<<(E * 4 + 255) / 256, blk, 0, stream>>>(ei, al, mb, sb, E);
    edge_scatter_kernel<<<gEdgeW, blk, 0, stream>>>(ei, al, sb, vb, xb, E);
    relu_kernel<<<gNode, blk, 0, stream>>>(xb, N * HO_C);

    // ---------------- edge scoring head ----------------
    edge_out_kernel<<<gEdgeW, blk, 0, stream>>>(ei, xb, wc, bc, out, E);
}
